// SAGE_62715112456383
// MI455X (gfx1250) — matrix operand order repaired and checked by reading, not compiled, not run on hardware
//
#include <hip/hip_runtime.h>
#include <hip/hip_fp16.h>

#define D 128
#define LN_EPS 1e-5f

typedef __attribute__((ext_vector_type(16))) _Float16 v16h;
typedef __attribute__((ext_vector_type(8)))  _Float16 v8h;
typedef __attribute__((ext_vector_type(8)))  float    v8f;

// ---------------- utility kernels ----------------

__global__ void sage_zero_f32(float* __restrict__ p, long n) {
    long i = (long)blockIdx.x * blockDim.x + threadIdx.x;
    if (i < n) p[i] = 0.0f;
}

__global__ void sage_deg(const int* __restrict__ dst, int E, float* __restrict__ deg) {
    int e = blockIdx.x * blockDim.x + threadIdx.x;
    if (e < E) atomicAdd(&deg[dst[e]], 1.0f);
}

__global__ void sage_inv(float* __restrict__ inv, int N) {  // in-place deg -> 1/(deg+1)
    int i = blockIdx.x * blockDim.x + threadIdx.x;
    if (i < N) inv[i] = 1.0f / (inv[i] + 1.0f);
}

// Wt[mat][n][k] = (f16) W[mat][k][n]   (column-major f16 so B fragments are contiguous)
__global__ void sage_convert_w(const float* __restrict__ W0, const float* __restrict__ W1,
                               const float* __restrict__ W2, _Float16* __restrict__ Wt) {
    int i = blockIdx.x * blockDim.x + threadIdx.x;
    if (i >= 3 * D * D) return;
    int mat = i / (D * D);
    int r   = i % (D * D);
    int n   = r / D;
    int k   = r % D;
    const float* W = (mat == 0) ? W0 : (mat == 1) ? W1 : W2;
    Wt[i] = (_Float16)W[k * D + n];
}

// scatter-add: one lane handles 4 consecutive floats of one edge's feature row
__global__ void sage_scatter(const int* __restrict__ src, const int* __restrict__ dst,
                             const float* __restrict__ h, float* __restrict__ agg, int E) {
    int t = blockIdx.x * blockDim.x + threadIdx.x;
    int e = t >> 5;
    int c = (t & 31) << 2;
    if (e >= E) return;
    const float4 v = *(const float4*)(h + (long)src[e] * D + c);
    float* a = agg + (long)dst[e] * D + c;
    atomicAdd(a + 0, v.x);
    atomicAdd(a + 1, v.y);
    atomicAdd(a + 2, v.z);
    atomicAdd(a + 3, v.w);
}

// ---------------- fused (agg+h)*inv -> x@W + b [-> LN -> ReLU] ----------------
// block = 256 threads = 8 waves; block handles a 16-node tile, wave w handles cols [16w,16w+16)

__global__ __launch_bounds__(256)
void sage_gemm_fused(const float* __restrict__ agg, const float* __restrict__ hin,
                     const float* __restrict__ inv, const _Float16* __restrict__ Wt,
                     const float* __restrict__ bias, const float* __restrict__ gamma,
                     const float* __restrict__ beta, float* __restrict__ out,
                     int N, int do_ln) {
    __shared__ __align__(16) _Float16 sA[16][D];   // x in f16, 4 KB
    __shared__ float sC[16][D];                    // fp32 result, 8 KB
    __shared__ float sRed[2][16][16];
    __shared__ float sMu[16], sRs[16];

    const int  tid  = threadIdx.x;
    const int  lane = tid & 31;
    const int  wave = tid >> 5;          // 0..7 -> 16-col block
    const long row0 = (long)blockIdx.x * 16;

    // ---- build x = (agg + h) * inv in f16 (2048 elems / 256 threads = 8 each) ----
    {
        int  base = tid * 8;
        int  m    = base / D;
        int  k    = base % D;
        long node = row0 + m;
        if (node < N) {
            float iv = inv[node];
            const float* ap = agg + node * D + k;
            const float* hp = hin + node * D + k;
            #pragma unroll
            for (int i = 0; i < 8; ++i)
                sA[m][k + i] = (_Float16)((ap[i] + hp[i]) * iv);
        } else {
            #pragma unroll
            for (int i = 0; i < 8; ++i) sA[m][k + i] = (_Float16)0.0f;
        }
    }
    __syncthreads();

    // ---- WMMA: D[16x16] = A[16x128] x B[128x16], 4 K-blocks of 32 ----
    const int mrow = lane & 15;
    const int hi   = lane >> 4;                // lane group
    const int col  = wave * 16 + mrow;
    v8f c = {};
    #pragma unroll
    for (int kb = 0; kb < 4; ++kb) {
        // A fragment: lanes0-15 K = 32kb+{0..7, 16..23}; lanes16-31 K = 32kb+{8..15, 24..31}
        const int ka0 = kb * 32 + hi * 8;
        const int ka1 = kb * 32 + 16 + hi * 8;
        v8h alo = *(const v8h*)(&sA[mrow][ka0]);
        v8h ahi = *(const v8h*)(&sA[mrow][ka1]);
        v16h a;
        #pragma unroll
        for (int i = 0; i < 8; ++i) { a[i] = alo[i]; a[i + 8] = ahi[i]; }

        // B fragment: lanes0-15 K = 32kb+{0..7, 16..23}; lanes16-31 K = 32kb+{8..15, 24..31}
        const _Float16* bp = Wt + (long)col * D + kb * 32 + hi * 8;
        v8h blo = *(const v8h*)(bp);
        v8h bhi = *(const v8h*)(bp + 16);
        v16h b;
        #pragma unroll
        for (int i = 0; i < 8; ++i) { b[i] = blo[i]; b[i + 8] = bhi[i]; }

        c = __builtin_amdgcn_wmma_f32_16x16x32_f16(false, a, false, b, (short)0, c,
                                                   false, false);
    }

    // ---- C fragment -> LDS (+bias). VGPR r of lane: row = r + 8*hi, col fixed per lane ----
    {
        float bv = bias[col];
        #pragma unroll
        for (int r = 0; r < 8; ++r)
            sC[r + hi * 8][col] = c[r] + bv;
    }
    __syncthreads();

    // ---- epilogue: LN+ReLU or passthrough; coalesced writeout ----
    {
        int  m   = tid >> 4;        // row 0..15
        int  seg = tid & 15;        // 8 cols per thread
        long node = row0 + m;
        if (do_ln) {
            float s = 0.0f, s2 = 0.0f;
            #pragma unroll
            for (int i = 0; i < 8; ++i) {
                float x = sC[m][seg * 8 + i];
                s += x; s2 += x * x;
            }
            sRed[0][m][seg] = s;
            sRed[1][m][seg] = s2;
            __syncthreads();
            if (seg == 0) {
                float ts = 0.0f, ts2 = 0.0f;
                #pragma unroll
                for (int j = 0; j < 16; ++j) { ts += sRed[0][m][j]; ts2 += sRed[1][m][j]; }
                float mu  = ts * (1.0f / D);
                float var = ts2 * (1.0f / D) - mu * mu;
                sMu[m] = mu;
                sRs[m] = rsqrtf(var + LN_EPS);
            }
            __syncthreads();
            if (node < N) {
                float mu = sMu[m], rs = sRs[m];
                #pragma unroll
                for (int i = 0; i < 8; ++i) {
                    int   k = seg * 8 + i;
                    float x = (sC[m][k] - mu) * rs * gamma[k] + beta[k];
                    out[node * D + k] = fmaxf(x, 0.0f);
                }
            }
        } else {
            if (node < N) {
                #pragma unroll
                for (int i = 0; i < 8; ++i) {
                    int k = seg * 8 + i;
                    out[node * D + k] = sC[m][k];
                }
            }
        }
    }
}

// ---------------- host side ----------------

extern "C" void kernel_launch(void* const* d_in, const int* in_sizes, int n_in,
                              void* d_out, int out_size, void* d_ws, size_t ws_size,
                              hipStream_t stream) {
    (void)n_in; (void)out_size; (void)ws_size;

    const float* feats = (const float*)d_in[0];
    const int*   esrc  = (const int*)d_in[1];
    const int*   edst  = (const int*)d_in[2];
    const float* W0    = (const float*)d_in[3];
    const float* b0    = (const float*)d_in[4];
    const float* W1    = (const float*)d_in[5];
    const float* b1    = (const float*)d_in[6];
    const float* W2    = (const float*)d_in[7];
    const float* b2    = (const float*)d_in[8];
    const float* g0    = (const float*)d_in[9];
    const float* be0   = (const float*)d_in[10];
    const float* g1    = (const float*)d_in[11];
    const float* be1   = (const float*)d_in[12];

    const int N = in_sizes[0] / D;
    const int E = in_sizes[1];

    // workspace carve-out (256B aligned)
    char*  ws  = (char*)d_ws;
    size_t off = 0;
    auto carve = [&](size_t bytes) {
        size_t o = off;
        off = (off + bytes + 255) & ~(size_t)255;
        return o;
    };
    float*    inv  = (float*)(ws + carve((size_t)N * sizeof(float)));
    _Float16* Wt   = (_Float16*)(ws + carve((size_t)3 * D * D * sizeof(_Float16)));
    float*    agg  = (float*)(ws + carve((size_t)N * D * sizeof(float)));
    float*    hbuf = (float*)(ws + carve((size_t)N * D * sizeof(float)));
    float*    outf = (float*)d_out;

    const int T = 256;
    const long ND = (long)N * D;

    // degrees -> inv
    sage_zero_f32<<<(int)((N + T - 1) / T), T, 0, stream>>>(inv, N);
    sage_deg<<<(E + T - 1) / T, T, 0, stream>>>(edst, E, inv);
    sage_inv<<<(N + T - 1) / T, T, 0, stream>>>(inv, N);

    // weights -> f16 column-major
    sage_convert_w<<<(3 * D * D + T - 1) / T, T, 0, stream>>>(W0, W1, W2, Wt);

    const float* Wl[3] = {nullptr, nullptr, nullptr}; (void)Wl;
    const int   gemm_blocks = (N + 15) / 16;
    const int   scat_blocks = (int)(((long)E * 32 + T - 1) / T);
    const int   zero_blocks = (int)((ND + T - 1) / T);

    for (int L = 0; L < 3; ++L) {
        const float* hin = (L == 0) ? feats : hbuf;
        float*       hout = (L == 2) ? outf : hbuf;   // in-place safe: per-tile read-then-write
        const _Float16* WtL = Wt + (size_t)L * D * D;
        const float* bL  = (L == 0) ? b0 : (L == 1) ? b1 : b2;
        const float* gL  = (L == 0) ? g0 : g1;        // unused when do_ln==0
        const float* beL = (L == 0) ? be0 : be1;

        sage_zero_f32<<<zero_blocks, T, 0, stream>>>(agg, ND);
        sage_scatter<<<scat_blocks, T, 0, stream>>>(esrc, edst, hin, agg, E);
        sage_gemm_fused<<<gemm_blocks, T, 0, stream>>>(agg, hin, inv, WtL, bL, gL, beL,
                                                       hout, N, (L < 2) ? 1 : 0);
    }
}
